// GroupedQueryAttentionDirect_7619271983135
// MI455X (gfx1250) — matrix operand order repaired and checked by reading, not compiled, not run on hardware
//
#include <hip/hip_runtime.h>
#include <hip/hip_bf16.h>

// GQA: S=2048, H=32 heads, G=8 KV groups, D=128, MODEL=4096
#define S_LEN 2048
#define NHEAD 32
#define NGRP  8
#define HDIM  128
#define HPG   (NHEAD / NGRP)
#define MODEL (NHEAD * HDIM)
#define KTILE 32

typedef __bf16 bf16x16 __attribute__((ext_vector_type(16)));
typedef float  f32x8   __attribute__((ext_vector_type(8)));
typedef unsigned v4u   __attribute__((ext_vector_type(4)));
typedef int      v4i   __attribute__((ext_vector_type(4)));
typedef int      v8i   __attribute__((ext_vector_type(8)));

#if defined(__has_builtin)
#if __has_builtin(__builtin_amdgcn_tensor_load_to_lds)
#define HAVE_TDM 1
#endif
#if __has_builtin(__builtin_amdgcn_s_wait_tensorcnt)
#define WAIT_TENSORCNT(n) __builtin_amdgcn_s_wait_tensorcnt(n)
#endif
#endif
#ifndef HAVE_TDM
#define HAVE_TDM 0
#endif
#ifndef WAIT_TENSORCNT
#define WAIT_TENSORCNT(n) asm volatile("s_wait_tensorcnt %0" ::"i"(n))
#endif

union Frag {
  unsigned u[8];
  bf16x16  v;
};

__device__ __forceinline__ f32x8 zero8() {
  f32x8 c;
#pragma unroll
  for (int i = 0; i < 8; ++i) c[i] = 0.0f;
  return c;
}

__device__ __forceinline__ unsigned pack_bf16(float lo, float hi) {
  unsigned a = __builtin_bit_cast(unsigned, lo);
  unsigned b = __builtin_bit_cast(unsigned, hi);
  return (a >> 16) | (b & 0xffff0000u);
}

__device__ __forceinline__ unsigned short f2bf(float x) {
  return (unsigned short)(__builtin_bit_cast(unsigned, x) >> 16);
}

// A fragment: 16x32 bf16 (MxK) from row-major source [rows][ld].
__device__ __forceinline__ void load_fragA(Frag& f, const unsigned short* src,
                                           int ld, int row0, int col0, int lane) {
  const int r = lane & 15, h = lane >> 4;
  const unsigned short* p = src + (size_t)(row0 + r) * ld + col0 + 8 * h;
#pragma unroll
  for (int j = 0; j < 4; ++j) f.u[j]     = *(const unsigned*)(p + 2 * j);
#pragma unroll
  for (int j = 0; j < 4; ++j) f.u[4 + j] = *(const unsigned*)(p + 16 + 2 * j);
}

// B fragment: 32(K)x16(N) where source is B^T row-major [N][ld over K].
__device__ __forceinline__ void load_fragB(Frag& f, const unsigned short* srcT,
                                           int ld, int n0, int k0, int lane) {
  const int r = lane & 15, h = lane >> 4;
  const unsigned short* p = srcT + (size_t)(n0 + r) * ld + k0 + 8 * h;
#pragma unroll
  for (int j = 0; j < 4; ++j) f.u[j]     = *(const unsigned*)(p + 2 * j);
#pragma unroll
  for (int j = 0; j < 4; ++j) f.u[4 + j] = *(const unsigned*)(p + 16 + 2 * j);
}

__device__ __forceinline__ f32x8 wmma_bf16(const Frag& a, const Frag& b, f32x8 c) {
  return __builtin_amdgcn_wmma_f32_16x16x32_bf16(false, a.v, false, b.v,
                                                 (short)0, c, false, false);
}

#if HAVE_TDM
// One 2D TDM tile load: tensor row-major [dim1][dim0] (dim0 contiguous, 2-byte
// elements), tile [tile_d1][tile_d0] at gaddr -> LDS row-major at lds byte offset.
__device__ __forceinline__ void tdm_load_2d(unsigned lds_byte_off, const void* gaddr,
                                            unsigned tensor_d0, unsigned tensor_d1,
                                            unsigned stride0,
                                            unsigned tile_d0, unsigned tile_d1) {
  unsigned long long ga = (unsigned long long)gaddr;
  v4u g0;
  g0[0] = 1u;                                          // count=1, user descriptor
  g0[1] = lds_byte_off;                                // D#.lds_addr
  g0[2] = (unsigned)(ga & 0xffffffffu);                // D#.global_addr[31:0]
  g0[3] = (unsigned)((ga >> 32) & 0x01ffffffu)         // D#.global_addr[56:32]
        | (2u << 30);                                  // D#.type = 2 ("image")
  v8i g1;
  g1[0] = 0x00010000;                                  // data_size=1 (2 bytes)
  g1[1] = (int)((tensor_d0 & 0xffffu) << 16);          // tensor_dim0[15:0]
  g1[2] = (int)((tensor_d0 >> 16) |
                ((tensor_d1 & 0xffffu) << 16));        // tensor_dim0 hi | tensor_dim1 lo
  g1[3] = (int)((tensor_d1 >> 16) | (tile_d0 << 16));  // tensor_dim1 hi | tile_dim0
  g1[4] = (int)(tile_d1 & 0xffffu);                    // tile_dim1 (tile_dim2 = 0)
  g1[5] = (int)stride0;                                // tensor_dim0_stride[31:0]
  g1[6] = 0;
  g1[7] = 0;
  v4i gz = {0, 0, 0, 0};
#if __clang_major__ >= 23
  v8i gz8 = {0, 0, 0, 0, 0, 0, 0, 0};
  __builtin_amdgcn_tensor_load_to_lds(g0, g1, gz, gz, gz8, 0);
#else
  __builtin_amdgcn_tensor_load_to_lds(g0, g1, gz, gz, 0);
#endif
}

__device__ __forceinline__ unsigned lds_off(const void* p) {
  return (unsigned)(unsigned long long)p;
}
#endif  // HAVE_TDM

// ---------------- prep kernels: fp32 -> bf16 with layout changes ----------------

__global__ void cvt_q_kernel(const float* __restrict__ q, unsigned short* __restrict__ Qb) {
  unsigned i = blockIdx.x * 256u + threadIdx.x;  // over H*S*D = 2^23
  int d = i & (HDIM - 1);
  int s = (i >> 7) & (S_LEN - 1);
  int h = i >> 18;
  const float scale = 0.08838834764831845f;  // 128^-0.5 folded into Q
  Qb[i] = f2bf(q[((size_t)s * NHEAD + h) * HDIM + d] * scale);
}

__global__ void cvt_k_kernel(const float* __restrict__ k, unsigned short* __restrict__ Kb) {
  unsigned i = blockIdx.x * 256u + threadIdx.x;  // over G*S*D = 2^21
  int d = i & (HDIM - 1);
  int s = (i >> 7) & (S_LEN - 1);
  int g = i >> 18;
  Kb[i] = f2bf(k[((size_t)s * NGRP + g) * HDIM + d]);
}

__global__ void cvt_v_kernel(const float* __restrict__ v, unsigned short* __restrict__ Vt) {
  unsigned i = blockIdx.x * 256u + threadIdx.x;  // over G*D*S = 2^21
  int s = i & (S_LEN - 1);
  int d = (i >> 11) & (HDIM - 1);
  int g = i >> 18;
  Vt[i] = f2bf(v[((size_t)s * NGRP + g) * HDIM + d]);
}

__global__ void cvt_w_kernel(const float* __restrict__ w, unsigned short* __restrict__ Wb) {
  unsigned i = blockIdx.x * 256u + threadIdx.x;  // over MODEL*MODEL = 2^24
  Wb[i] = f2bf(w[i]);
}

// ---------------- flash-attention kernel (TDM double-buffered K/V, bf16 WMMA) ---------

__global__ void __launch_bounds__(128) gqa_attn_kernel(
    const unsigned short* __restrict__ Qb,  // [H][S][D] bf16 (pre-scaled)
    const unsigned short* __restrict__ Kb,  // [G][S][D] bf16
    const unsigned short* __restrict__ Vt,  // [G][D][S] bf16
    unsigned short* __restrict__ X) {       // [S][MODEL] bf16
#if HAVE_TDM
  __shared__ unsigned short ldsK[2][KTILE * HDIM];  // [keys][d]   8KB x2
  __shared__ unsigned short ldsV[2][HDIM * KTILE];  // [d][keys]   8KB x2
#endif
  __shared__ float plds[4][16 * 32];  // per-wave P-transpose staging

  const int lane = threadIdx.x & 31;
  const int wave = threadIdx.x >> 5;
  const int r = lane & 15, hh = lane >> 4;
  const int head = blockIdx.y;
  const int grp  = head / HPG;
  const int q0   = blockIdx.x * 64 + wave * 16;

  const unsigned short* Qh = Qb + (size_t)head * S_LEN * HDIM;
  const unsigned short* Kg = Kb + (size_t)grp * S_LEN * HDIM;
  const unsigned short* Vg = Vt + (size_t)grp * HDIM * S_LEN;

  Frag qf[4];
#pragma unroll
  for (int kd = 0; kd < 4; ++kd) load_fragA(qf[kd], Qh, HDIM, q0, kd * 32, lane);

  f32x8 acc[8];
#pragma unroll
  for (int d = 0; d < 8; ++d) acc[d] = zero8();
  float m[8], l[8];
#pragma unroll
  for (int v = 0; v < 8; ++v) { m[v] = -1e30f; l[v] = 0.0f; }

  float* pl = plds[wave];

#if HAVE_TDM
  const bool leader = (wave == 0);
  if (leader) {
    tdm_load_2d(lds_off(&ldsK[0][0]), Kg, HDIM, S_LEN, HDIM, HDIM, KTILE);
    tdm_load_2d(lds_off(&ldsV[0][0]), Vg, S_LEN, HDIM, S_LEN, KTILE, HDIM);
  }
#endif

#pragma unroll 1
  for (int it = 0; it < S_LEN / KTILE; ++it) {
    const int k0 = it * KTILE;
#if HAVE_TDM
    const int cur = it & 1;
    const int nxt = cur ^ 1;
    if (leader) {
      if (k0 + KTILE < S_LEN) {
        tdm_load_2d(lds_off(&ldsK[nxt][0]), Kg + (size_t)(k0 + KTILE) * HDIM,
                    HDIM, S_LEN, HDIM, HDIM, KTILE);
        tdm_load_2d(lds_off(&ldsV[nxt][0]), Vg + (k0 + KTILE),
                    S_LEN, HDIM, S_LEN, KTILE, HDIM);
        WAIT_TENSORCNT(2);  // tile `it`'s 2 in-order TDM ops complete
      } else {
        WAIT_TENSORCNT(0);
      }
    }
    __syncthreads();  // publish LDS tile `it`
    const unsigned short* Ksrc = &ldsK[cur][0]; const int kld = HDIM;  const int kb = 0;
    const unsigned short* Vsrc = &ldsV[cur][0]; const int vld = KTILE; const int vb = 0;
#else
    const unsigned short* Ksrc = Kg; const int kld = HDIM;  const int kb = k0;
    const unsigned short* Vsrc = Vg; const int vld = S_LEN; const int vb = k0;
#endif

    // ---- scores: batch-load all 8 K B-fragments, then chain 8 WMMAs so the
    //      scheduler can overlap ds_load latency with the matrix pipe.
    Frag kfb[2][4];
#pragma unroll
    for (int n = 0; n < 2; ++n)
#pragma unroll
      for (int kd = 0; kd < 4; ++kd)
        load_fragB(kfb[n][kd], Ksrc, kld, kb + n * 16, kd * 32, lane);

    f32x8 sc[2];
#pragma unroll
    for (int n = 0; n < 2; ++n) {
      f32x8 c = zero8();
#pragma unroll
      for (int kd = 0; kd < 4; ++kd) c = wmma_bf16(qf[kd], kfb[n][kd], c);
      sc[n] = c;
    }

    // ---- online softmax
    float alpha[8];
#pragma unroll
    for (int v = 0; v < 8; ++v) {
      float mx = fmaxf(sc[0][v], sc[1][v]);
#pragma unroll
      for (int o = 8; o >= 1; o >>= 1) mx = fmaxf(mx, __shfl_xor(mx, o, 16));
      float mn = fmaxf(m[v], mx);
      alpha[v] = __expf(m[v] - mn);
      m[v] = mn;
      float p0 = __expf(sc[0][v] - mn);
      float p1 = __expf(sc[1][v] - mn);
      sc[0][v] = p0; sc[1][v] = p1;
      float rs = p0 + p1;
#pragma unroll
      for (int o = 8; o >= 1; o >>= 1) rs += __shfl_xor(rs, o, 16);
      l[v] = l[v] * alpha[v] + rs;
    }
#pragma unroll
    for (int d = 0; d < 8; ++d)
#pragma unroll
      for (int v = 0; v < 8; ++v) acc[d][v] *= alpha[v];

    // ---- transpose P 16x32 through LDS into an A-fragment
#pragma unroll
    for (int v = 0; v < 8; ++v) {
      pl[(v + 8 * hh) * 32 + r]      = sc[0][v];
      pl[(v + 8 * hh) * 32 + 16 + r] = sc[1][v];
    }
    Frag pf;
    {
      const float* pr = pl + r * 32 + 8 * hh;
#pragma unroll
      for (int j = 0; j < 4; ++j) pf.u[j]     = pack_bf16(pr[2 * j],      pr[2 * j + 1]);
#pragma unroll
      for (int j = 0; j < 4; ++j) pf.u[4 + j] = pack_bf16(pr[16 + 2 * j], pr[16 + 2 * j + 1]);
    }

    // ---- acc += P(16x32) @ V(32x128): batch-load 8 V fragments, then 8 WMMAs
    Frag vfb[8];
#pragma unroll
    for (int d = 0; d < 8; ++d) load_fragB(vfb[d], Vsrc, vld, d * 16, vb, lane);
#pragma unroll
    for (int d = 0; d < 8; ++d) acc[d] = wmma_bf16(pf, vfb[d], acc[d]);

#if HAVE_TDM
    __syncthreads();  // all waves done with buffer `cur` before refill
#endif
  }

  // ---- finalize
  float inv[8];
#pragma unroll
  for (int v = 0; v < 8; ++v) inv[v] = 1.0f / l[v];
#pragma unroll
  for (int d = 0; d < 8; ++d) {
#pragma unroll
    for (int v = 0; v < 8; ++v) {
      int q = q0 + v + 8 * hh;
      int col = head * HDIM + d * 16 + r;
      X[(size_t)q * MODEL + col] = f2bf(acc[d][v] * inv[v]);
    }
  }
}

// ---------------- output projection: out = X @ W^T + b (bf16 WMMA, f32 out) -----------

__global__ void __launch_bounds__(128) out_proj_kernel(
    const unsigned short* __restrict__ X,   // [S][MODEL] bf16
    const unsigned short* __restrict__ Wb,  // [MODEL][MODEL] bf16 (row m = out feature)
    const float* __restrict__ bias,         // [MODEL]
    float* __restrict__ out) {              // [S][MODEL] f32
  const int lane = threadIdx.x & 31;
  const int wave = threadIdx.x >> 5;
  const int r = lane & 15, hh = lane >> 4;
  const int q0 = blockIdx.x * 32;
  const int n0 = blockIdx.y * 512 + wave * 128;

  f32x8 acc[2][8];
#pragma unroll
  for (int mt = 0; mt < 2; ++mt)
#pragma unroll
    for (int nt = 0; nt < 8; ++nt) acc[mt][nt] = zero8();

#pragma unroll 1
  for (int k0 = 0; k0 < MODEL; k0 += 32) {
    if (k0 + 32 < MODEL) {
      __builtin_prefetch(X + (size_t)(q0 + (lane & 15)) * MODEL + k0 + 32, 0, 1);
      __builtin_prefetch(Wb + (size_t)(n0 + (lane & 15)) * MODEL + k0 + 32, 0, 1);
    }
    // batch-load all fragments for this K step, then run the 16 WMMAs
    Frag a0, a1, bfr[8];
    load_fragA(a0, X, MODEL, q0,      k0, lane);
    load_fragA(a1, X, MODEL, q0 + 16, k0, lane);
#pragma unroll
    for (int nt = 0; nt < 8; ++nt)
      load_fragB(bfr[nt], Wb, MODEL, n0 + nt * 16, k0, lane);
#pragma unroll
    for (int nt = 0; nt < 8; ++nt) {
      acc[0][nt] = wmma_bf16(a0, bfr[nt], acc[0][nt]);
      acc[1][nt] = wmma_bf16(a1, bfr[nt], acc[1][nt]);
    }
  }

#pragma unroll
  for (int mt = 0; mt < 2; ++mt)
#pragma unroll
    for (int nt = 0; nt < 8; ++nt)
#pragma unroll
      for (int v = 0; v < 8; ++v) {
        int q = q0 + mt * 16 + v + 8 * hh;
        int n = n0 + nt * 16 + r;
        out[(size_t)q * MODEL + n] = acc[mt][nt][v] + bias[n];
      }
}

// ---------------- launch ----------------

extern "C" void kernel_launch(void* const* d_in, const int* in_sizes, int n_in,
                              void* d_out, int out_size, void* d_ws, size_t ws_size,
                              hipStream_t stream) {
  const float* q = (const float*)d_in[0];  // [S,H,D]
  const float* k = (const float*)d_in[1];  // [S,G,D]
  const float* v = (const float*)d_in[2];  // [S,G,D]
  // d_in[3] = mask: unused (reference never applies it)
  const float* w = (const float*)d_in[4];  // [MODEL,MODEL]
  const float* b = (const float*)d_in[5];  // [MODEL]
  float* out = (float*)d_out;              // [S,H,D] == [S,MODEL]

  // workspace: Qb 16MiB | Kb 4MiB | Vt 4MiB | Wb 32MiB | X 16MiB = 72MiB
  char* ws = (char*)d_ws;
  unsigned short* Qb = (unsigned short*)(ws);
  unsigned short* Kb = (unsigned short*)(ws + 16777216);
  unsigned short* Vt = (unsigned short*)(ws + 20971520);
  unsigned short* Wb = (unsigned short*)(ws + 25165824);
  unsigned short* X  = (unsigned short*)(ws + 58720256);

  cvt_q_kernel<<<(NHEAD * S_LEN * HDIM) / 256, 256, 0, stream>>>(q, Qb);
  cvt_k_kernel<<<(NGRP * S_LEN * HDIM) / 256, 256, 0, stream>>>(k, Kb);
  cvt_v_kernel<<<(NGRP * S_LEN * HDIM) / 256, 256, 0, stream>>>(v, Vt);
  cvt_w_kernel<<<(MODEL * MODEL) / 256, 256, 0, stream>>>(w, Wb);

  gqa_attn_kernel<<<dim3(S_LEN / 64, NHEAD), 128, 0, stream>>>(Qb, Kb, Vt, X);
  out_proj_kernel<<<dim3(S_LEN / 32, MODEL / 512), 128, 0, stream>>>(X, Wb, b, out);
}
